// Attention_28492813042193
// MI455X (gfx1250) — matrix operand order repaired and checked by reading, not compiled, not run on hardware
//
#include <hip/hip_runtime.h>
#include <hip/hip_bf16.h>

// ---------------------------------------------------------------------------
// Dual-softmax cross-attention for MI455X (gfx1250), wave32 + WMMA f16.
//   q  = (x1 @ Wq) * SCALE ; q2 = (x2 @ Wq2) * SCALE  (scale folded into proj)
//   k  = context @ Wk ; v = context @ Wv (stored transposed [b,h,d,m])
//   attn = 0.3*softmax(q k^T) + 0.7*softmax(q2 k^T)   (flash-fused, 2 states)
//   out  = (attn @ v) @ Wo + bo
// All GEMMs via v_wmma_f32_16x16x32_f16.  Weights pre-packed into B-fragment
// layout.  Attention K/V tiles staged in LDS via async global->LDS copies
// (double-buffered, ASYNCcnt + workgroup barrier protocol).
// ---------------------------------------------------------------------------

#define NHEADS 4
#define DHEAD  64
#define QDIM   512
#define CDIM   512
#define INNER  256
#define BATCH  4
#define SEQ    2048
#define SCALEF 0.125f   // 64^-0.5

typedef __attribute__((ext_vector_type(16))) _Float16 v16h;
typedef __attribute__((ext_vector_type(8)))  _Float16 v8h;
typedef __attribute__((ext_vector_type(8)))  float    v8f;
typedef __attribute__((ext_vector_type(4)))  int      v4i;

__device__ __forceinline__ v16h join8(v8h lo, v8h hi) {
  v16h r;
#pragma unroll
  for (int i = 0; i < 8; ++i) { r[i] = lo[i]; r[i + 8] = hi[i]; }
  return r;
}

__device__ __forceinline__ v8f wmma16(v16h a, v16h b, v8f c) {
  // D = A(16x32 f16) * B(32x16 f16) + C(16x16 f32)
  return __builtin_amdgcn_wmma_f32_16x16x32_f16(false, a, false, b, (short)0, c,
                                                false, false);
}

// ---- async global -> LDS (16B per lane), ASYNCcnt-tracked ------------------
#if __has_builtin(__builtin_amdgcn_global_load_async_to_lds_b128)
#define HAS_ASYNC_LDS 1
#else
#define HAS_ASYNC_LDS 0
#endif

__device__ __forceinline__ void cp16_async(const _Float16* g, _Float16* s) {
#if HAS_ASYNC_LDS
  __builtin_amdgcn_global_load_async_to_lds_b128(
      (__attribute__((address_space(1))) v4i*)g,
      (__attribute__((address_space(3))) v4i*)s, 0, 0);
#else
  *(v8h*)s = *(const v8h*)g;   // fallback: load + ds_store
#endif
}

__device__ __forceinline__ void async_wait_all() {
#if HAS_ASYNC_LDS
#if __has_builtin(__builtin_amdgcn_s_wait_asynccnt)
  __builtin_amdgcn_s_wait_asynccnt(0);
#else
  asm volatile("s_wait_asynccnt 0x0" ::: "memory");
#endif
#endif
}

// ---------------------------------------------------------------------------
// Weight re-pack: W_f32[K,N] -> Wt_f16[(kt*N + n)*32 + kk] with k = kt*32+kk.
// A lane loading B-fragment (k0, col n) then reads 32B contiguous:
//   Wt + ((k0>>5)*N + n)*32 + grp*16
// ---------------------------------------------------------------------------
__global__ __launch_bounds__(256) void prep_w(const float* __restrict__ W,
                                              _Float16* __restrict__ Wt,
                                              int K, int N) {
  const int idx = blockIdx.x * 256 + threadIdx.x;     // = (kt*N + n)*32 + kk
  if (idx >= K * N) return;
  const int kk = idx & 31;
  const int rest = idx >> 5;
  const int n  = rest % N;
  const int kt = rest / N;
  Wt[idx] = (_Float16)W[(kt * 32 + kk) * N + n];
}

// ---------------------------------------------------------------------------
// Projection: Y_f16 = (X_f32[8192,512] @ W[512,256]) * outScale
// Grid 8192/64, block 128 (4 waves).  Each wave: 16 rows x all 256 cols.
// B fragments double-buffered so loads overlap WMMA issue.
// ---------------------------------------------------------------------------
__global__ __launch_bounds__(128) void proj_kernel(
    const float* __restrict__ X, const _Float16* __restrict__ Wt,
    _Float16* __restrict__ Y, int storeT, float outScale) {
  const int lane = threadIdx.x & 31;
  const int wave = threadIdx.x >> 5;
  const int ln   = lane & 15;
  const int grp  = lane >> 4;                 // 0 or 1
  const int m0   = blockIdx.x * 64 + wave * 16;

  v8f acc[16] = {};
  for (int k0 = 0; k0 < QDIM; k0 += 32) {
    // A fragment (16x32): lane = M row, halves K = grp*16..+15 (contiguous)
    v16h a;
    const float* ap = X + (m0 + ln) * QDIM + k0 + grp * 16;
#pragma unroll
    for (int i = 0; i < 8; ++i) a[i] = (_Float16)ap[i];
#pragma unroll
    for (int i = 0; i < 8; ++i) a[i + 8] = (_Float16)ap[8 + i];

    const _Float16* wb = Wt + ((size_t)(k0 >> 5) * INNER + ln) * 32 + grp * 16;
    v16h bf = *(const v16h*)wb;
#pragma unroll
    for (int nt = 0; nt < 16; ++nt) {
      const v16h cb = bf;
      if (nt + 1 < 16) bf = *(const v16h*)(wb + (size_t)(nt + 1) * 16 * 32);
      acc[nt] = wmma16(a, cb, acc[nt]);
    }
  }

#pragma unroll
  for (int nt = 0; nt < 16; ++nt) {
    const int c = nt * 16 + ln;               // global column (lane = N)
    const int h = c >> 6, d = c & 63;
    const int rbase = m0 + grp * 8;           // rows rbase..rbase+7
    const int b = rbase >> 11, n = rbase & 2047;
    if (!storeT) {
#pragma unroll
      for (int i = 0; i < 8; ++i)
        Y[(((size_t)b * NHEADS + h) * SEQ + (n + i)) * DHEAD + d] =
            (_Float16)(acc[nt][i] * outScale);
    } else {
      v8h st;
#pragma unroll
      for (int i = 0; i < 8; ++i) st[i] = (_Float16)(acc[nt][i] * outScale);
      *(v8h*)&Y[(((size_t)b * NHEADS + h) * DHEAD + d) * SEQ + n] = st;
    }
  }
}

// ---------------------------------------------------------------------------
// Online (flash) update for one score pair: s[2] holds two 16x16 f32 score
// tiles (C layout).  Updates running max/sum, rescales O, converts P to f16
// A-layout via per-wave LDS transpose, and accumulates O += P @ V.
// ---------------------------------------------------------------------------
__device__ __forceinline__ void flash_update(v8f (&s)[2], v8f (&O)[4],
                                             float (&mm)[8], float (&l)[8],
                                             const v16h (&vb)[4],
                                             _Float16* sb, int ln, int grp) {
  float al[8];
#pragma unroll
  for (int i = 0; i < 8; ++i) {
    float v = fmaxf(s[0][i], s[1][i]);
    v = fmaxf(v, __shfl_xor(v, 1, 32));
    v = fmaxf(v, __shfl_xor(v, 2, 32));
    v = fmaxf(v, __shfl_xor(v, 4, 32));
    v = fmaxf(v, __shfl_xor(v, 8, 32));      // rows stay within 16-lane group
    const float mn = fmaxf(mm[i], v);
    al[i] = __expf(mm[i] - mn);
    mm[i] = mn;
  }
#pragma unroll
  for (int i = 0; i < 8; ++i) {
    const float p0 = __expf(s[0][i] - mm[i]);
    const float p1 = __expf(s[1][i] - mm[i]);
    float sm = p0 + p1;
    sm += __shfl_xor(sm, 1, 32);
    sm += __shfl_xor(sm, 2, 32);
    sm += __shfl_xor(sm, 4, 32);
    sm += __shfl_xor(sm, 8, 32);
    l[i] = l[i] * al[i] + sm;
#pragma unroll
    for (int nt = 0; nt < 4; ++nt) O[nt][i] *= al[i];
    // C layout -> LDS row-major 16x32 (row = grp*8+i, col = jc*16+ln)
    sb[(grp * 8 + i) * 32 + ln]      = (_Float16)p0;
    sb[(grp * 8 + i) * 32 + 16 + ln] = (_Float16)p1;
  }
  // Re-read in A layout: lane row = ln, K halves grp*16..+15 (contiguous)
  const _Float16* pp = sb + ln * 32 + grp * 16;
  const v16h pa = join8(*(const v8h*)pp, *(const v8h*)(pp + 8));
#pragma unroll
  for (int nt = 0; nt < 4; ++nt) O[nt] = wmma16(pa, vb[nt], O[nt]);
}

// ---------------------------------------------------------------------------
// Fused dual-softmax attention.  Grid (SEQ/64, NHEADS, BATCH), block 128.
// Each wave owns 16 query rows.  K/V j-tiles (32 wide) are staged in LDS by
// async global->LDS copies, double-buffered across iterations:
//   iter i: wait own ASYNCcnt -> barrier (all fills visible, all readers of
//   the other buffer done) -> launch fill of next tile -> compute from LDS.
// ---------------------------------------------------------------------------
__global__ __launch_bounds__(128) void attn_kernel(
    const _Float16* __restrict__ Qf, const _Float16* __restrict__ Q2f,
    const _Float16* __restrict__ Kf, const _Float16* __restrict__ Vt,
    _Float16* __restrict__ Of) {
  __shared__ _Float16 Ksh[2][32 * 64];   // [j=32][d=64]
  __shared__ _Float16 Vsh[2][64 * 32];   // [d=64][j=32]  (from V^T)
  __shared__ _Float16 Psh[4][16 * 32];   // per-wave P transpose scratch
  const int tid  = threadIdx.x;
  const int lane = tid & 31;
  const int wave = tid >> 5;
  const int ln   = lane & 15;
  const int grp  = lane >> 4;
  const int bh   = blockIdx.z * NHEADS + blockIdx.y;
  const int m0   = blockIdx.x * 64 + wave * 16;

  const _Float16* Qb  = Qf  + (size_t)bh * SEQ * DHEAD;
  const _Float16* Q2b = Q2f + (size_t)bh * SEQ * DHEAD;
  const _Float16* Kb  = Kf  + (size_t)bh * SEQ * DHEAD;
  const _Float16* Vb  = Vt  + (size_t)bh * DHEAD * SEQ;
  _Float16* sb = Psh[wave];

  // Q fragments (A layout), two K=32 chunks over d=64, for both queries.
  v16h qa[2], q2a[2];
#pragma unroll
  for (int kk = 0; kk < 2; ++kk) {
    const _Float16* p = Qb + (m0 + ln) * DHEAD + kk * 32 + grp * 16;
    qa[kk] = join8(*(const v8h*)p, *(const v8h*)(p + 8));
    const _Float16* p2 = Q2b + (m0 + ln) * DHEAD + kk * 32 + grp * 16;
    q2a[kk] = join8(*(const v8h*)p2, *(const v8h*)(p2 + 8));
  }

  v8f O1[4] = {}, O2[4] = {};
  float m1[8], m2[8], l1[8] = {}, l2[8] = {};
#pragma unroll
  for (int i = 0; i < 8; ++i) { m1[i] = -3.0e38f; m2[i] = -3.0e38f; }

  // Cooperative tile fill: 128 threads x 2 chunks x 16B cover K (4KB) and V
  // (4KB).  K slice [jt..jt+31][0..63] is contiguous in global memory; V^T
  // rows have stride SEQ.
  auto fill = [&](int buf, int jt) {
#pragma unroll
    for (int c = 0; c < 2; ++c) {
      const int q = tid + c * 128;                 // 0..255 chunk id
      cp16_async(Kb + (size_t)jt * DHEAD + q * 8, &Ksh[buf][q * 8]);
      const int d = q >> 2, part = q & 3;          // V: 4 chunks per d-row
      cp16_async(Vb + (size_t)d * SEQ + jt + part * 8,
                 &Vsh[buf][d * 32 + part * 8]);
    }
  };

  fill(0, 0);
  for (int jt = 0, it = 0; jt < SEQ; jt += 32, ++it) {
    const int cur = it & 1;
    async_wait_all();      // own async fills of Ksh/Vsh[cur] complete
    __syncthreads();       // everyone's fills visible; prev readers done
    if (jt + 32 < SEQ) fill(cur ^ 1, jt + 32);

    // K^T B-fragments from LDS: lane = j col, halves = d (contiguous)
    v16h kb[2][2];
#pragma unroll
    for (int jc = 0; jc < 2; ++jc)
#pragma unroll
      for (int kk = 0; kk < 2; ++kk)
        kb[jc][kk] = *(const v16h*)(&Ksh[cur][(jc * 16 + ln) * DHEAD +
                                              kk * 32 + grp * 16]);
    // V B-fragments from LDS: lane = d col, halves = j (contiguous)
    v16h vb[4];
#pragma unroll
    for (int nt = 0; nt < 4; ++nt)
      vb[nt] = *(const v16h*)(&Vsh[cur][(nt * 16 + ln) * 32 + grp * 16]);

    v8f s1[2] = {}, s2[2] = {};
#pragma unroll
    for (int jc = 0; jc < 2; ++jc) {
      s1[jc] = wmma16(qa[0],  kb[jc][0], s1[jc]);
      s1[jc] = wmma16(qa[1],  kb[jc][1], s1[jc]);
      s2[jc] = wmma16(q2a[0], kb[jc][0], s2[jc]);
      s2[jc] = wmma16(q2a[1], kb[jc][1], s2[jc]);
    }
    flash_update(s1, O1, m1, l1, vb, sb, ln, grp);
    flash_update(s2, O2, m2, l2, vb, sb, ln, grp);
  }

  // out = 0.3*O1/l1 + 0.7*O2/l2  ->  f16 [b,h,n,d]
#pragma unroll
  for (int nt = 0; nt < 4; ++nt)
#pragma unroll
    for (int i = 0; i < 8; ++i) {
      const float v = 0.3f * O1[nt][i] / l1[i] + 0.7f * O2[nt][i] / l2[i];
      const int r = m0 + grp * 8 + i;
      Of[((size_t)bh * SEQ + r) * DHEAD + nt * 16 + ln] = (_Float16)v;
    }
}

// ---------------------------------------------------------------------------
// Output projection: out_f32[8192,512] = Of_f16[b,h,n,d] @ Wo[256,512] + bo
// Grid (8192/64, 512/128), block 128.  Wave: 16 rows x 128 cols.
// ---------------------------------------------------------------------------
__global__ __launch_bounds__(128) void out_proj(
    const _Float16* __restrict__ Of, const _Float16* __restrict__ Wot,
    const float* __restrict__ bo, float* __restrict__ out) {
  const int lane = threadIdx.x & 31;
  const int wave = threadIdx.x >> 5;
  const int ln   = lane & 15;
  const int grp  = lane >> 4;
  const int m0   = blockIdx.x * 64 + wave * 16;
  const int n0   = blockIdx.y * 128;

  const int t = m0 + ln;                 // A row = token (lane = M)
  const int b = t >> 11, n = t & 2047;

  v8f acc[8] = {};
  for (int k0 = 0; k0 < INNER; k0 += 32) {
    const int h = k0 >> 6, d0 = k0 & 63;  // k-tile stays inside one head
    const _Float16* ap =
        Of + (((size_t)b * NHEADS + h) * SEQ + n) * DHEAD + d0 + grp * 16;
    const v16h a = join8(*(const v8h*)ap, *(const v8h*)(ap + 8));
    const _Float16* wb =
        Wot + ((size_t)(k0 >> 5) * QDIM + n0 + ln) * 32 + grp * 16;
    v16h bf = *(const v16h*)wb;
#pragma unroll
    for (int nt = 0; nt < 8; ++nt) {
      const v16h cb = bf;
      if (nt + 1 < 8) bf = *(const v16h*)(wb + (size_t)(nt + 1) * 16 * 32);
      acc[nt] = wmma16(a, cb, acc[nt]);
    }
  }
#pragma unroll
  for (int nt = 0; nt < 8; ++nt) {
    const int c = n0 + nt * 16 + ln;
    const float bias = bo[c];
#pragma unroll
    for (int i = 0; i < 8; ++i) {
      const int r = m0 + grp * 8 + i;
      out[(size_t)r * QDIM + c] = acc[nt][i] + bias;
    }
  }
}

// ---------------------------------------------------------------------------
extern "C" void kernel_launch(void* const* d_in, const int* in_sizes, int n_in,
                              void* d_out, int out_size, void* d_ws,
                              size_t ws_size, hipStream_t stream) {
  const float* x1  = (const float*)d_in[0];
  const float* x2  = (const float*)d_in[1];
  const float* ctx = (const float*)d_in[2];
  const float* Wq  = (const float*)d_in[3];
  const float* Wq2 = (const float*)d_in[4];
  const float* Wk  = (const float*)d_in[5];
  const float* Wv  = (const float*)d_in[6];
  const float* Wo  = (const float*)d_in[7];
  const float* bo  = (const float*)d_in[8];
  float* out = (float*)d_out;

  const size_t elems = (size_t)BATCH * NHEADS * SEQ * DHEAD;  // 2M f16 = 4MB
  const size_t welems = (size_t)QDIM * INNER;                 // 128K f16
  _Float16* Qf   = (_Float16*)d_ws;
  _Float16* Q2f  = Qf + elems;
  _Float16* Kf   = Q2f + elems;
  _Float16* Vt   = Kf + elems;   // transposed [b,h,d,m]
  _Float16* Of   = Vt + elems;
  _Float16* Wqt  = Of + elems;
  _Float16* Wq2t = Wqt + welems;
  _Float16* Wkt  = Wq2t + welems;
  _Float16* Wvt  = Wkt + welems;
  _Float16* Wot  = Wvt + welems;  // [256,512] repacked

  // Re-pack weights into B-fragment-native f16 layout (L2 resident).
  const int wblk = (int)(welems / 256);
  prep_w<<<wblk, 256, 0, stream>>>(Wq,  Wqt,  QDIM, INNER);
  prep_w<<<wblk, 256, 0, stream>>>(Wq2, Wq2t, QDIM, INNER);
  prep_w<<<wblk, 256, 0, stream>>>(Wk,  Wkt,  CDIM, INNER);
  prep_w<<<wblk, 256, 0, stream>>>(Wv,  Wvt,  CDIM, INNER);
  prep_w<<<wblk, 256, 0, stream>>>(Wo,  Wot,  INNER, QDIM);

  dim3 gp((BATCH * SEQ) / 64, 1, 1);
  proj_kernel<<<gp, 128, 0, stream>>>(x1,  Wqt,  Qf,  0, SCALEF);
  proj_kernel<<<gp, 128, 0, stream>>>(x2,  Wq2t, Q2f, 0, SCALEF);
  proj_kernel<<<gp, 128, 0, stream>>>(ctx, Wkt,  Kf,  0, 1.0f);
  proj_kernel<<<gp, 128, 0, stream>>>(ctx, Wvt,  Vt,  1, 1.0f);

  dim3 ga(SEQ / 64, NHEADS, BATCH);
  attn_kernel<<<ga, 128, 0, stream>>>(Qf, Q2f, Kf, Vt, Of);

  dim3 go((BATCH * SEQ) / 64, QDIM / 128, 1);
  out_proj<<<go, 128, 0, stream>>>(Of, Wot, bo, out);
}
